// LM_67095979098175
// MI455X (gfx1250) — matrix operand order repaired and checked by reading, not compiled, not run on hardware
//
#include <hip/hip_runtime.h>
#include <cstddef>

// ---- problem constants (match reference) ----
#define VOCAB 128
#define E     512
#define NL    8
#define NH    8
#define BB    4
#define TT    1024
#define BT    (BB * TT)      // 4096 tokens
#define DH    64             // head dim
#define QKVW  (3 * E)        // 1536

typedef __attribute__((ext_vector_type(16))) __bf16 v16bf;
typedef __attribute__((ext_vector_type(8)))  float  v8f;
typedef int v4i __attribute__((vector_size(16)));   // matches builtin param type

// ---------------------------------------------------------------------------
// WMMA helper (wave32, CDNA5 ISA 7.12)
// ---------------------------------------------------------------------------
__device__ __forceinline__ v8f wmma_bf16(v16bf a, v16bf b, v8f c) {
  return __builtin_amdgcn_wmma_f32_16x16x32_bf16(
      false, a, false, b, (short)0, c, false, false);
}

// A fragment: 16x32 (MxK), fp32 row-major source, convert to bf16.
// lane g=lane>>4, m=lane&15 ; element i -> K=kBase+g*16+i
__device__ __forceinline__ v16bf load_a_f32(const float* A, int lda,
                                            int mBase, int kBase, int lane) {
  int g = lane >> 4, m = lane & 15;
  const float* p = A + (size_t)(mBase + m) * lda + kBase + g * 16;
  v16bf a;
#pragma unroll
  for (int i = 0; i < 8; ++i) a[i] = (__bf16)p[i];
#pragma unroll
  for (int i = 0; i < 8; ++i) a[i + 8] = (__bf16)p[8 + i];
  return a;
}

// B fragment from LDS panel laid out [n][32k] bf16 (k contiguous):
// lane n=lane&15 -> column nOff+n ; K = (lane>>4)*16 + i  (32B contiguous read)
__device__ __forceinline__ v16bf load_b_lds32(const __bf16* L, int nOff, int lane) {
  int g = lane >> 4, n = lane & 15;
  return *(const v16bf*)(L + (size_t)(nOff + n) * 32 + g * 16);
}

// ---- async global->LDS staging (pure 16B copy), with safe fallback ----
__device__ __forceinline__ void stage16B(__bf16* ldst, const __bf16* gsrc) {
#if defined(__AMDGCN__) && __has_builtin(__builtin_amdgcn_global_load_async_to_lds_b128)
  __builtin_amdgcn_global_load_async_to_lds_b128(
      (__attribute__((address_space(1))) v4i*)(gsrc),
      (__attribute__((address_space(3))) v4i*)(ldst), 0, 0);
#else
  *(float4*)ldst = *(const float4*)gsrc;
#endif
}
__device__ __forceinline__ void stage_wait() {
#if defined(__AMDGCN__) && __has_builtin(__builtin_amdgcn_global_load_async_to_lds_b128)
#if __has_builtin(__builtin_amdgcn_s_wait_asynccnt)
  __builtin_amdgcn_s_wait_asynccnt(0);
#else
  asm volatile("s_wait_asynccnt 0x0" ::: "memory");
#endif
#endif
}

// ---- wave reductions (wave32) ----
__device__ __forceinline__ float wave_sum(float v) {
#pragma unroll
  for (int off = 16; off >= 1; off >>= 1) v += __shfl_xor(v, off, 32);
  return v;
}
__device__ __forceinline__ float wave_max(float v) {
#pragma unroll
  for (int off = 16; off >= 1; off >>= 1) v = fmaxf(v, __shfl_xor(v, off, 32));
  return v;
}

// ---------------------------------------------------------------------------
// GEMM: C[M,N] = act(A[M,K] @ W[N,K]^T + bias), fp32 A, bf16 W.
// Block = 256 thr = 8 waves; block tile 256 rows x 64 cols (wave: 32x64).
// B panel (64n x 32k bf16, 4KB) staged in LDS, double-buffered, async copy.
// grid = (N/64, M/256). K multiple of 32.
// ---------------------------------------------------------------------------
__global__ void k_gemm(const float* __restrict__ A, int lda,
                       const __bf16* __restrict__ W, int ldw,
                       const float* __restrict__ bias,
                       float* __restrict__ C, int ldc,
                       int K, int relu) {
  __shared__ __bf16 ldsB[2][64 * 32];
  int tid = threadIdx.x;
  int lane = tid & 31, w = tid >> 5;
  int mBase = blockIdx.y * 256 + w * 32;
  int nBase = blockIdx.x * 64;
  int sn = tid >> 2, sk = (tid & 3) * 8;   // staging: 16B per thread
  const __bf16* wsrc = W + (size_t)(nBase + sn) * ldw + sk;

  v8f acc[2][4] = {};
  stage16B(&ldsB[0][sn * 32 + sk], wsrc);
  int buf = 0;
  for (int k = 0; k < K; k += 32) {
    stage_wait();
    __syncthreads();
    if (k + 32 < K) stage16B(&ldsB[buf ^ 1][sn * 32 + sk], wsrc + (k + 32));
    v16bf a0 = load_a_f32(A, lda, mBase, k, lane);
    v16bf a1 = load_a_f32(A, lda, mBase + 16, k, lane);
#pragma unroll
    for (int t = 0; t < 4; ++t) {
      v16bf b = load_b_lds32(&ldsB[buf][0], t * 16, lane);
      acc[0][t] = wmma_bf16(a0, b, acc[0][t]);
      acc[1][t] = wmma_bf16(a1, b, acc[1][t]);
    }
    buf ^= 1;
  }
  int g = lane >> 4, nn = lane & 15;
#pragma unroll
  for (int mt = 0; mt < 2; ++mt)
#pragma unroll
    for (int t = 0; t < 4; ++t) {
      int col = nBase + t * 16 + nn;
      float bv = bias ? bias[col] : 0.0f;
#pragma unroll
      for (int r = 0; r < 8; ++r) {
        float vv = acc[mt][t][r] + bv;
        if (relu) vv = fmaxf(vv, 0.0f);
        C[(size_t)(mBase + mt * 16 + g * 8 + r) * ldc + col] = vv;
      }
    }
}

// ---------------------------------------------------------------------------
// Attention scores: scores[p,m,n] = (q_m . k_n) * 0.125 * (n<=m).
// Block tile 128 rows x 64 cols; K-panel (64n x 64k) staged once in LDS (bf16).
// Whole-masked blocks early-out before any barrier (condition uniform/block).
// ---------------------------------------------------------------------------
__global__ void k_attn_scores(const float* __restrict__ qkv,
                              float* __restrict__ scores, int bh0) {
  __shared__ __bf16 ldsK[64 * 64];   // [n][k], 8KB
  int tid = threadIdx.x, lane = tid & 31, w = tid >> 5;
  int p = blockIdx.z, bh = bh0 + p, b = bh >> 3, h = bh & 7;
  int mBase0 = blockIdx.y * 128;
  int mBase = mBase0 + w * 16;
  int nBase = blockIdx.x * 64;
  int g = lane >> 4, nn = lane & 15;
  float* out = scores + (size_t)p * TT * TT;

  if (nBase >= mBase0 + 128) {        // entire block above diagonal
#pragma unroll
    for (int t = 0; t < 4; ++t)
#pragma unroll
      for (int r = 0; r < 8; ++r)
        out[(size_t)(mBase + g * 8 + r) * TT + nBase + t * 16 + nn] = 0.0f;
    return;
  }

  const float* q = qkv + (size_t)b * TT * QKVW + h * (3 * DH);
  const float* kkp = q + DH;
  {  // stage K panel: 4 threads per n-row, 16 contiguous k each
    int n = tid >> 2, k0 = (tid & 3) * 16;
    const float* src = kkp + (size_t)(nBase + n) * QKVW + k0;
    __bf16* dst = &ldsK[n * 64 + k0];
#pragma unroll
    for (int i = 0; i < 16; ++i) dst[i] = (__bf16)src[i];
  }
  __syncthreads();

  v8f acc[4] = {};
  if (nBase < mBase + 16) {           // wave touches the kept triangle
    for (int k = 0; k < DH; k += 32) {
      v16bf a = load_a_f32(q, QKVW, mBase, k, lane);
#pragma unroll
      for (int t = 0; t < 4; ++t) {
        v16bf bfrag = *(const v16bf*)&ldsK[(size_t)(t * 16 + nn) * 64 + k + g * 16];
        acc[t] = wmma_bf16(a, bfrag, acc[t]);
      }
    }
  }
#pragma unroll
  for (int t = 0; t < 4; ++t)
#pragma unroll
    for (int r = 0; r < 8; ++r) {
      int m = mBase + g * 8 + r, n = nBase + t * 16 + nn;
      float v = (n <= m) ? acc[t][r] * 0.125f : 0.0f;
      out[(size_t)m * TT + n] = v;
    }
}

// Row softmax over TT columns; one wave per row (32 values per lane).
__global__ void k_softmax(float* __restrict__ scores, int nrows) {
  int wid = (blockIdx.x * blockDim.x + threadIdx.x) >> 5;
  int lane = threadIdx.x & 31;
  if (wid >= nrows) return;
  float* row = scores + (size_t)wid * TT;
  float v[32];
  float mx = -3.4e38f;
#pragma unroll
  for (int j = 0; j < 32; ++j) { v[j] = row[lane + j * 32]; mx = fmaxf(mx, v[j]); }
  mx = wave_max(mx);
  float s = 0.0f;
#pragma unroll
  for (int j = 0; j < 32; ++j) { v[j] = __expf(v[j] - mx); s += v[j]; }
  s = wave_sum(s);
  float inv = 1.0f / s;
#pragma unroll
  for (int j = 0; j < 32; ++j) row[lane + j * 32] = v[j] * inv;
}

// ---------------------------------------------------------------------------
// att = P @ V, written into attout[BT,E] at column h*64.
// V chunk (32k x 64d) staged in LDS transposed to [d][k] bf16 (4KB),
// double-buffered; B fragments become contiguous 32B ds reads.
// grid = (1, TT/128, CH)
// ---------------------------------------------------------------------------
__global__ void k_attn_av(const float* __restrict__ qkv,
                          const float* __restrict__ scores,
                          float* __restrict__ attout, int bh0) {
  __shared__ __bf16 ldsV[2][64 * 32];  // [d][k]
  int tid = threadIdx.x, lane = tid & 31, w = tid >> 5;
  int p = blockIdx.z, bh = bh0 + p, b = bh >> 3, h = bh & 7;
  int mBase = blockIdx.y * 128 + w * 16;
  const float* P = scores + (size_t)p * TT * TT;
  const float* V = qkv + (size_t)b * TT * QKVW + h * (3 * DH) + 2 * DH;
  int skk = tid >> 3, sd0 = (tid & 7) * 8;   // staging coords

  v8f acc[4] = {};
  {  // stage k-chunk 0
    const float* src = V + (size_t)skk * QKVW + sd0;
#pragma unroll
    for (int j = 0; j < 8; ++j) ldsV[0][(size_t)(sd0 + j) * 32 + skk] = (__bf16)src[j];
  }
  int buf = 0;
  for (int k = 0; k < TT; k += 32) {
    __syncthreads();
    if (k + 32 < TT) {
      const float* src = V + (size_t)(k + 32 + skk) * QKVW + sd0;
#pragma unroll
      for (int j = 0; j < 8; ++j)
        ldsV[buf ^ 1][(size_t)(sd0 + j) * 32 + skk] = (__bf16)src[j];
    }
    v16bf a = load_a_f32(P, TT, mBase, k, lane);
#pragma unroll
    for (int t = 0; t < 4; ++t) {
      v16bf bfrag = load_b_lds32(&ldsV[buf][0], t * 16, lane);
      acc[t] = wmma_bf16(a, bfrag, acc[t]);
    }
    buf ^= 1;
  }
  int g = lane >> 4, nn = lane & 15;
#pragma unroll
  for (int t = 0; t < 4; ++t)
#pragma unroll
    for (int r = 0; r < 8; ++r) {
      int m = mBase + g * 8 + r;
      attout[(size_t)(b * TT + m) * E + h * DH + t * 16 + nn] = acc[t][r];
    }
}

// out = LayerNorm(A + Bres) * s + b ; one wave per token (E=512 -> 16/lane).
__global__ void k_add_ln(const float* __restrict__ A, const float* __restrict__ Bres,
                         const float* __restrict__ s, const float* __restrict__ bb,
                         float* __restrict__ out) {
  int wid = (blockIdx.x * blockDim.x + threadIdx.x) >> 5;
  int lane = threadIdx.x & 31;
  const float* a = A + (size_t)wid * E;
  const float* rz = Bres + (size_t)wid * E;
  float v[16];
  float sum = 0.0f;
#pragma unroll
  for (int j = 0; j < 16; ++j) { v[j] = a[lane + j * 32] + rz[lane + j * 32]; sum += v[j]; }
  float mean = wave_sum(sum) * (1.0f / E);
  float vs = 0.0f;
#pragma unroll
  for (int j = 0; j < 16; ++j) { float d = v[j] - mean; vs += d * d; }
  float var = wave_sum(vs) * (1.0f / E);
  float inv = rsqrtf(var + 1e-5f);
#pragma unroll
  for (int j = 0; j < 16; ++j) {
    int e = lane + j * 32;
    out[(size_t)wid * E + e] = (v[j] - mean) * inv * s[e] + bb[e];
  }
}

// embedding gather
__global__ void k_embed(const long long* __restrict__ inp,
                        const float* __restrict__ emb, float* __restrict__ x) {
  int i = blockIdx.x * 256 + threadIdx.x;            // BT*E threads
  int t = i >> 9, e = i & 511;
  x[i] = emb[(size_t)((int)inp[t]) * E + e];
}

// fp32 -> bf16 weight conversion
__global__ void k_cvt(const float* __restrict__ src, __bf16* __restrict__ dst, int n) {
  int i = blockIdx.x * 256 + threadIdx.x;
  if (i < n) dst[i] = (__bf16)src[i];
}

// per-token loss: (1-EPS)*nll + EPS*smooth ; one wave per token (V=128 -> 4/lane)
__global__ void k_token_loss(const float* __restrict__ logits,
                             const long long* __restrict__ tgt,
                             float* __restrict__ tl) {
  int wid = (blockIdx.x * blockDim.x + threadIdx.x) >> 5;
  int lane = threadIdx.x & 31;
  const float* lg = logits + (size_t)wid * VOCAB;
  float v[4];
  float mx = -3.4e38f, tot = 0.0f;
#pragma unroll
  for (int j = 0; j < 4; ++j) { v[j] = lg[lane + j * 32]; mx = fmaxf(mx, v[j]); tot += v[j]; }
  mx = wave_max(mx);
  tot = wave_sum(tot);
  float se = 0.0f;
#pragma unroll
  for (int j = 0; j < 4; ++j) se += __expf(v[j] - mx);
  se = wave_sum(se);
  if (lane == 0) {
    float lse = mx + __logf(se);
    float nll = lse - lg[(int)tgt[wid]];
    float smooth = lse - tot * (1.0f / VOCAB);
    tl[wid] = 0.9f * nll + 0.1f * smooth;
  }
}

__global__ void k_reduce_loss(const float* __restrict__ tl, float* __restrict__ out) {
  __shared__ float sm[256];
  float s = 0.0f;
  for (int i = threadIdx.x; i < BT; i += 256) s += tl[i];
  sm[threadIdx.x] = s;
  __syncthreads();
  for (int st = 128; st > 0; st >>= 1) {
    if ((int)threadIdx.x < st) sm[threadIdx.x] += sm[threadIdx.x + st];
    __syncthreads();
  }
  if (threadIdx.x == 0) out[0] = sm[0] * (1.0f / BT);
}

// ---------------------------------------------------------------------------
extern "C" void kernel_launch(void* const* d_in, const int* in_sizes, int n_in,
                              void* d_out, int out_size, void* d_ws, size_t ws_size,
                              hipStream_t stream) {
  (void)in_sizes; (void)n_in; (void)out_size;
  const long long* inputs  = (const long long*)d_in[0];
  const long long* targets = (const long long*)d_in[1];
  const float* emb    = (const float*)d_in[2];
  const float* qkv_w  = (const float*)d_in[3];
  const float* qkv_b  = (const float*)d_in[4];
  const float* aff1_w = (const float*)d_in[5];
  const float* aff1_b = (const float*)d_in[6];
  const float* aff2_w = (const float*)d_in[7];
  const float* aff2_b = (const float*)d_in[8];
  const float* ffn1_w = (const float*)d_in[9];
  const float* ffn1_b = (const float*)d_in[10];
  const float* ffn2_w = (const float*)d_in[11];
  const float* ffn2_b = (const float*)d_in[12];
  const float* ln1_s  = (const float*)d_in[13];
  const float* ln1_b  = (const float*)d_in[14];
  const float* ln2_s  = (const float*)d_in[15];
  const float* ln2_b  = (const float*)d_in[16];
  const float* head_w = (const float*)d_in[17];
  const float* head_b = (const float*)d_in[18];

  // ---- workspace partition (256B aligned) ----
  char* wsp = (char*)d_ws;
  auto alloc = [&](size_t bytes) -> char* {
    char* p = wsp;
    wsp += (bytes + 255) & ~(size_t)255;
    return p;
  };
  const int nQKV = NL * QKVW * E;
  const int nSQ  = NL * E * E;
  const int nHD  = VOCAB * E;
  __bf16* wb_qkv  = (__bf16*)alloc((size_t)nQKV * 2);
  __bf16* wb_a1   = (__bf16*)alloc((size_t)nSQ * 2);
  __bf16* wb_a2   = (__bf16*)alloc((size_t)nSQ * 2);
  __bf16* wb_f1   = (__bf16*)alloc((size_t)nSQ * 2);
  __bf16* wb_f2   = (__bf16*)alloc((size_t)nSQ * 2);
  __bf16* wb_head = (__bf16*)alloc((size_t)nHD * 2);
  float* x      = (float*)alloc((size_t)BT * E * 4);
  float* qkvb   = (float*)alloc((size_t)BT * QKVW * 4);
  float* attout = (float*)alloc((size_t)BT * E * 4);
  float* t1     = (float*)alloc((size_t)BT * E * 4);
  float* t2     = (float*)alloc((size_t)BT * E * 4);
  float* out1   = (float*)alloc((size_t)BT * E * 4);
  float* tl     = (float*)alloc((size_t)BT * 4);
  size_t used = (size_t)(wsp - (char*)d_ws);
  size_t avail = ws_size > used ? ws_size - used : 0;
  int CH = 32;
  while (CH > 1 && (size_t)CH * TT * TT * 4 > avail) CH >>= 1;
  float* scores = (float*)alloc((size_t)CH * TT * TT * 4);

  // ---- weight conversion ----
  k_cvt<<<(nQKV + 255) / 256, 256, 0, stream>>>(qkv_w, wb_qkv, nQKV);
  k_cvt<<<(nSQ + 255) / 256, 256, 0, stream>>>(aff1_w, wb_a1, nSQ);
  k_cvt<<<(nSQ + 255) / 256, 256, 0, stream>>>(aff2_w, wb_a2, nSQ);
  k_cvt<<<(nSQ + 255) / 256, 256, 0, stream>>>(ffn1_w, wb_f1, nSQ);
  k_cvt<<<(nSQ + 255) / 256, 256, 0, stream>>>(ffn2_w, wb_f2, nSQ);
  k_cvt<<<(nHD + 255) / 256, 256, 0, stream>>>(head_w, wb_head, nHD);

  // ---- embedding ----
  k_embed<<<(BT * E) / 256, 256, 0, stream>>>(inputs, emb, x);

  dim3 blk(256);
  dim3 gE(E / 64, BT / 256);
  for (int l = 0; l < NL; ++l) {
    // qkv = x @ qkv_w^T + qkv_b
    k_gemm<<<dim3(QKVW / 64, BT / 256), blk, 0, stream>>>(
        x, E, wb_qkv + (size_t)l * QKVW * E, E, qkv_b + (size_t)l * QKVW,
        qkvb, QKVW, E, 0);

    // attention over (b,h) chunks
    for (int c0 = 0; c0 < BB * NH; c0 += CH) {
      k_attn_scores<<<dim3(TT / 64, TT / 128, CH), blk, 0, stream>>>(qkvb, scores, c0);
      k_softmax<<<(CH * TT) / 8, blk, 0, stream>>>(scores, CH * TT);
      k_attn_av<<<dim3(1, TT / 128, CH), blk, 0, stream>>>(qkvb, scores, attout, c0);
    }

    // att-affine
    k_gemm<<<gE, blk, 0, stream>>>(attout, E, wb_a1 + (size_t)l * E * E, E,
                                   aff1_b + (size_t)l * E, t1, E, E, 1);
    k_gemm<<<gE, blk, 0, stream>>>(t1, E, wb_a2 + (size_t)l * E * E, E,
                                   aff2_b + (size_t)l * E, t2, E, E, 0);
    k_add_ln<<<BT / 8, blk, 0, stream>>>(t2, x, ln1_s + (size_t)l * E,
                                         ln1_b + (size_t)l * E, out1);
    // ffn
    k_gemm<<<gE, blk, 0, stream>>>(out1, E, wb_f1 + (size_t)l * E * E, E,
                                   ffn1_b + (size_t)l * E, t1, E, E, 1);
    k_gemm<<<gE, blk, 0, stream>>>(t1, E, wb_f2 + (size_t)l * E * E, E,
                                   ffn2_b + (size_t)l * E, t2, E, E, 0);
    k_add_ln<<<BT / 8, blk, 0, stream>>>(t2, out1, ln2_s + (size_t)l * E,
                                         ln2_b + (size_t)l * E, x);
  }

  // head: logits (first BT*V floats of d_out)
  float* logits = (float*)d_out;
  k_gemm<<<dim3(VOCAB / 64, BT / 256), blk, 0, stream>>>(
      x, E, wb_head, E, head_b, logits, VOCAB, E, 0);

  // loss -> d_out[BT*V]
  k_token_loss<<<BT / 8, blk, 0, stream>>>(logits, targets, tl);
  k_reduce_loss<<<1, blk, 0, stream>>>(tl, logits + (size_t)BT * VOCAB);
}
